// CustomGPT2Block_65231963291956
// MI455X (gfx1250) — matrix operand order repaired and checked by reading, not compiled, not run on hardware
//
#include <hip/hip_runtime.h>

typedef __attribute__((ext_vector_type(16))) _Float16 v16h;
typedef __attribute__((ext_vector_type(8)))  _Float16 v8h;
typedef __attribute__((ext_vector_type(8)))  float    v8f;

// Problem constants (match reference)
constexpr int Bq  = 2;
constexpr int S   = 2048;
constexpr int D   = 768;
constexpr int H   = 12;
constexpr int DH  = 64;
constexpr int HID = 1536;
constexpr int MT  = Bq * S;            // 4096 token rows
constexpr float EPS = 1e-6f;

// -------------------- fragment helpers (ISA 7.12.2 layouts) --------------------

__device__ __forceinline__ v16h frag_cat(v8h lo, v8h hi) {
  return __builtin_shufflevector(lo, hi, 0,1,2,3,4,5,6,7,8,9,10,11,12,13,14,15);
}

// A fragment (16x32 f16): lane holds row m=lane&15; halves i: k = (lane>>4)*8 + (i&7) + (i>=8?16:0)
// source is row-major [row][k], ld in elements. Two contiguous 16B loads per lane.
__device__ __forceinline__ v16h load_fragA(const _Float16* base, int ld, int row, int k0, int lane) {
  const _Float16* p = base + (size_t)row * ld + k0 + ((lane >> 4) << 3);
  v8h lo = *(const v8h*)(p);
  v8h hi = *(const v8h*)(p + 16);
  return frag_cat(lo, hi);
}

// B fragment (32x16 f16): lane holds col n=lane&15; halves i: k = (lane>>4)*8 + (i&7) + (i>=8?16:0)
// source is row-major-by-n [n][k] (i.e. B transposed), ld in elements. Two contiguous 16B loads.
__device__ __forceinline__ v16h load_fragB(const _Float16* base, int ld, int row, int k0, int lane) {
  const _Float16* p = base + (size_t)row * ld + k0 + ((lane >> 4) << 3);
  v8h lo = *(const v8h*)(p);
  v8h hi = *(const v8h*)(p + 16);
  return frag_cat(lo, hi);
}

__device__ __forceinline__ v8f wmma16(v16h a, v16h b, v8f c) {
  return __builtin_amdgcn_wmma_f32_16x16x32_f16(false, a, false, b, (short)0, c, false, false);
}

__device__ __forceinline__ v8f zero8() {
  v8f z = {0.f,0.f,0.f,0.f,0.f,0.f,0.f,0.f};
  return z;
}

// -------------------- weight convert: fp32 [K][N] -> f16 transposed [N][K] --------------------

__global__ void wcvt_transpose(const float* __restrict__ W, _Float16* __restrict__ WT, int K, int N) {
  int idx = blockIdx.x * blockDim.x + threadIdx.x;
  if (idx >= K * N) return;
  int k = idx / N;
  int n = idx - k * N;
  WT[(size_t)n * K + k] = (_Float16)W[idx];
}

// -------------------- RMSNorm (fp32 in) -> f16 out, one row per block --------------------

__global__ void rmsnorm_f16(const float* __restrict__ x, const float* __restrict__ w,
                            _Float16* __restrict__ out) {
  __shared__ float red[256];
  int row = blockIdx.x;
  int tid = threadIdx.x;
  const float* xr = x + (size_t)row * D;
  float v0 = xr[tid];
  float v1 = xr[tid + 256];
  float v2 = xr[tid + 512];
  red[tid] = v0 * v0 + v1 * v1 + v2 * v2;
  __syncthreads();
  for (int sft = 128; sft > 0; sft >>= 1) {
    if (tid < sft) red[tid] += red[tid + sft];
    __syncthreads();
  }
  float scale = rsqrtf(red[0] * (1.0f / (float)D) + EPS);
  _Float16* orow = out + (size_t)row * D;
  orow[tid]       = (_Float16)(v0 * scale * w[tid]);
  orow[tid + 256] = (_Float16)(v1 * scale * w[tid + 256]);
  orow[tid + 512] = (_Float16)(v2 * scale * w[tid + 512]);
}

// -------------------- generic WMMA GEMM, wave tile 32x64, block tile 64x256 --------------------
// C[M,N] = A[M,K] (f16 row-major) * B (f16, stored transposed: BT[N][K]) + bias
// EPI 0: QKV scatter  (q/k -> [B,H,S,DH], v -> [B,H,DH,S])
// EPI 1: ReLU -> f16 out [M,N]
// EPI 2: + residual(f32) -> f32 out [M,N]

#define EPI_QKV    0
#define EPI_RELU16 1
#define EPI_RES32  2

template <int EPI>
__global__ void gemm_wmma(const _Float16* __restrict__ A, const _Float16* __restrict__ BT,
                          const float* __restrict__ bias,
                          const float* res, float* out32, _Float16* out16,
                          _Float16* q16, _Float16* k16, _Float16* vT16,
                          int M, int N, int K) {
  int lane = threadIdx.x & 31;
  int wave = threadIdx.x >> 5;       // 0..7
  int wm = wave >> 2;                // 0..1
  int wn = wave & 3;                 // 0..3
  int m0 = blockIdx.x * 64 + wm * 32;
  int n0 = blockIdx.y * 256 + wn * 64;

  v8f acc[2][4];
#pragma unroll
  for (int i = 0; i < 2; ++i)
#pragma unroll
    for (int t = 0; t < 4; ++t) acc[i][t] = zero8();

  int rowA = m0 + (lane & 15);
  for (int k0 = 0; k0 < K; k0 += 32) {
    v16h a0 = load_fragA(A, K, rowA,      k0, lane);
    v16h a1 = load_fragA(A, K, rowA + 16, k0, lane);
#pragma unroll
    for (int t = 0; t < 4; ++t) {
      v16h b = load_fragB(BT, K, n0 + t * 16 + (lane & 15), k0, lane);
      acc[0][t] = wmma16(a0, b, acc[0][t]);
      acc[1][t] = wmma16(a1, b, acc[1][t]);
    }
  }

  int mloc = (lane >> 4) << 3;       // 0 or 8
  int nlan = lane & 15;
#pragma unroll
  for (int mi = 0; mi < 2; ++mi) {
#pragma unroll
    for (int t = 0; t < 4; ++t) {
      int n = n0 + t * 16 + nlan;
      float bn = bias[n];
#pragma unroll
      for (int r = 0; r < 8; ++r) {
        int m = m0 + mi * 16 + mloc + r;
        float h = acc[mi][t][r] + bn;
        if (EPI == EPI_RELU16) {
          out16[(size_t)m * N + n] = (_Float16)fmaxf(h, 0.0f);
        } else if (EPI == EPI_RES32) {
          out32[(size_t)m * N + n] = h + res[(size_t)m * N + n];
        } else { // QKV scatter
          int which = n / D;               // 0=q 1=k 2=v
          int dm    = n - which * D;
          int head  = dm >> 6;             // /64
          int dh    = dm & 63;
          int b     = m >> 11;             // /S (2048)
          int s     = m & (S - 1);
          _Float16 hv = (_Float16)h;
          size_t hb = (size_t)(b * H + head);
          if (which == 0)      q16 [(hb * S + s) * DH + dh] = hv;
          else if (which == 1) k16 [(hb * S + s) * DH + dh] = hv;
          else                 vT16[(hb * DH + dh) * S + s] = hv;
        }
      }
    }
  }
}

// -------------------- attention: squared-ReLU, flash-style, 1 wave per 16-query tile ---------
// q16/k16: [B,H,S,DH] f16 row-major ; vT16: [B,H,DH,S] f16 ; ctx16 out: [B*S, D] f16

__global__ void attn_sqrelu(const _Float16* __restrict__ q16, const _Float16* __restrict__ k16,
                            const _Float16* __restrict__ vT16, _Float16* __restrict__ ctx16) {
  __shared__ _Float16 lds_probs[4][16 * 32];   // per-wave 16x32 probs staging
  int lane = threadIdx.x & 31;
  int wave = threadIdx.x >> 5;                 // 0..3
  int gidx = blockIdx.x * 4 + wave;            // 0 .. B*H*(S/16)-1 = 3071
  int qt = gidx & (S / 16 - 1);                // q-tile within (b,h)
  int bh = gidx >> 7;                          // 0..23
  const _Float16* qh = q16  + (size_t)bh * S * DH;
  const _Float16* kh = k16  + (size_t)bh * S * DH;
  const _Float16* vh = vT16 + (size_t)bh * DH * S;

  int qrow = qt * 16 + (lane & 15);
  v16h aq0 = load_fragA(qh, DH, qrow, 0,  lane);
  v16h aq1 = load_fragA(qh, DH, qrow, 32, lane);

  v8f c[4];
#pragma unroll
  for (int t = 0; t < 4; ++t) c[t] = zero8();

  _Float16* lp = lds_probs[wave];
  int mloc = (lane >> 4) << 3;
  int nlan = lane & 15;

  for (int kb = 0; kb < S; kb += 32) {
    // two 16x16 score tiles for this 32-key block
#pragma unroll
    for (int t = 0; t < 2; ++t) {
      v8f sc = zero8();
      int key = kb + t * 16 + nlan;
      v16h bk0 = load_fragB(kh, DH, key, 0,  lane);
      v16h bk1 = load_fragB(kh, DH, key, 32, lane);
      sc = wmma16(aq0, bk0, sc);
      sc = wmma16(aq1, bk1, sc);
      int col = t * 16 + nlan;
#pragma unroll
      for (int r = 0; r < 8; ++r) {
        float p = fmaxf(sc[r] * 0.125f, 0.0f);   // 1/sqrt(64)
        p = p * p;                                // squared ReLU
        lp[(mloc + r) * 32 + col] = (_Float16)p;
      }
    }
    // re-layout probs C->A fragment via per-wave LDS (wave-private, in-order DS)
    v16h ap = load_fragA(lp, 32, (lane & 15), 0, lane);
#pragma unroll
    for (int t2 = 0; t2 < 4; ++t2) {
      v16h bv = load_fragB(vh, S, t2 * 16 + nlan, kb, lane);
      c[t2] = wmma16(ap, bv, c[t2]);
    }
  }

  int b = bh / H, h = bh - (bh / H) * H;
#pragma unroll
  for (int t2 = 0; t2 < 4; ++t2) {
#pragma unroll
    for (int r = 0; r < 8; ++r) {
      int tok = b * S + qt * 16 + mloc + r;
      int d   = h * DH + t2 * 16 + nlan;
      ctx16[(size_t)tok * D + d] = (_Float16)c[t2][r];
    }
  }
}

// -------------------- launcher --------------------

extern "C" void kernel_launch(void* const* d_in, const int* in_sizes, int n_in,
                              void* d_out, int out_size, void* d_ws, size_t ws_size,
                              hipStream_t stream) {
  (void)in_sizes; (void)n_in; (void)out_size; (void)ws_size;
  const float* x      = (const float*)d_in[0];
  const float* ln1_w  = (const float*)d_in[1];
  const float* W_attn = (const float*)d_in[2];
  const float* b_attn = (const float*)d_in[3];
  const float* W_proj = (const float*)d_in[4];
  const float* b_proj = (const float*)d_in[5];
  const float* ln2_w  = (const float*)d_in[6];
  const float* W_fc1  = (const float*)d_in[7];
  const float* b_fc1  = (const float*)d_in[8];
  const float* W_fc2  = (const float*)d_in[9];
  const float* b_fc2  = (const float*)d_in[10];
  float* out = (float*)d_out;

  // workspace layout (f16 elements)
  _Float16* base = (_Float16*)d_ws;
  size_t off = 0;
  _Float16* wattnT = base + off; off += (size_t)(3 * D) * D;   // [2304][768]
  _Float16* wprojT = base + off; off += (size_t)D * D;          // [768][768]
  _Float16* wfc1T  = base + off; off += (size_t)HID * D;        // [1536][768]
  _Float16* wfc2T  = base + off; off += (size_t)D * HID;        // [768][1536]
  _Float16* h16    = base + off; off += (size_t)MT * D;
  _Float16* q16    = base + off; off += (size_t)Bq * H * S * DH;
  _Float16* k16    = base + off; off += (size_t)Bq * H * S * DH;
  _Float16* vT16   = base + off; off += (size_t)Bq * H * DH * S;
  _Float16* ctx16  = base + off; off += (size_t)MT * D;
  _Float16* h216   = base + off; off += (size_t)MT * D;
  _Float16* act16  = base + off; off += (size_t)MT * HID;

  // 1) weight conversion/transpose (all fit in L2 afterwards)
  {
    int n;
    n = D * 3 * D; wcvt_transpose<<<(n + 255) / 256, 256, 0, stream>>>(W_attn, wattnT, D, 3 * D);
    n = D * D;     wcvt_transpose<<<(n + 255) / 256, 256, 0, stream>>>(W_proj, wprojT, D, D);
    n = D * HID;   wcvt_transpose<<<(n + 255) / 256, 256, 0, stream>>>(W_fc1,  wfc1T,  D, HID);
    n = HID * D;   wcvt_transpose<<<(n + 255) / 256, 256, 0, stream>>>(W_fc2,  wfc2T,  HID, D);
  }

  // 2) RMSNorm 1 -> f16
  rmsnorm_f16<<<MT, 256, 0, stream>>>(x, ln1_w, h16);

  // 3) QKV projection with head-scatter epilogue (v stored transposed)
  gemm_wmma<EPI_QKV><<<dim3(MT / 64, (3 * D) / 256), 256, 0, stream>>>(
      h16, wattnT, b_attn, nullptr, nullptr, nullptr, q16, k16, vT16, MT, 3 * D, D);

  // 4) squared-ReLU attention -> ctx f16 [B*S, D]
  attn_sqrelu<<<(Bq * H * (S / 16)) / 4, 128, 0, stream>>>(q16, k16, vT16, ctx16);

  // 5) output projection + bias + residual(x) -> d_out (fp32)
  gemm_wmma<EPI_RES32><<<dim3(MT / 64, D / 256), 256, 0, stream>>>(
      ctx16, wprojT, b_proj, x, out, nullptr, nullptr, nullptr, nullptr, MT, D, D);

  // 6) RMSNorm 2 on d_out -> f16
  rmsnorm_f16<<<MT, 256, 0, stream>>>(out, ln2_w, h216);

  // 7) FC1 + bias + ReLU -> f16
  gemm_wmma<EPI_RELU16><<<dim3(MT / 64, HID / 256), 256, 0, stream>>>(
      h216, wfc1T, b_fc1, nullptr, nullptr, act16, nullptr, nullptr, nullptr, MT, HID, D);

  // 8) FC2 + bias + residual(d_out) -> d_out (each element read+written by the same lane)
  gemm_wmma<EPI_RES32><<<dim3(MT / 64, D / 256), 256, 0, stream>>>(
      act16, wfc2T, b_fc2, out, out, nullptr, nullptr, nullptr, nullptr, MT, D, HID);
}
